// Attention_1726576854632
// MI455X (gfx1250) — matrix operand order repaired and checked by reading, not compiled, not run on hardware
//
#include <hip/hip_runtime.h>
#include <math.h>

// Problem constants (fixed by the reference)
#define DIM     4096
#define NH      32
#define HD      128
#define BSZ     16
#define MAXSEQ  4096
#define NEWPOS  4095               // start_pos = MAX_SEQ - 1
#define KSPLIT  8                  // K-split for QKV GEMMs
#define KRANGE  (DIM / KSPLIT)     // 512
#define OSPLIT  8                  // K-split for output GEMM
#define ORANGE  (DIM / OSPLIT)     // 512
#define SPLITS  8                  // seq splits for flash decode
#define CHUNK   (MAXSEQ / SPLITS)  // 512
#define PSTRIDE 132                // per-(bh,split) partial: [m, l, pad, pad, acc[128]]

typedef __bf16 bf16_t;
typedef bf16_t v16bf __attribute__((ext_vector_type(16)));
typedef float  v8f   __attribute__((ext_vector_type(8)));

// ---------------------------------------------------------------------------
// A-fragment for V_WMMA_F32_16X16X32_BF16 (16x32, M = lane&15):
// lanes 0-15 hold K = 0..15 (elems 0..15);
// lanes 16-31 hold K = 16..31.
// ---------------------------------------------------------------------------
__device__ __forceinline__ v16bf make_a_frag(const float* __restrict__ xrow,
                                             int k0, int kg) {
  const float4* p0 = reinterpret_cast<const float4*>(xrow + k0 + kg * 16);
  const float4* p1 = reinterpret_cast<const float4*>(xrow + k0 + kg * 16 + 8);
  float4 a0 = p0[0], a1 = p0[1], a2 = p1[0], a3 = p1[1];
  v16bf a;
  a[0]  = (bf16_t)a0.x; a[1]  = (bf16_t)a0.y; a[2]  = (bf16_t)a0.z; a[3]  = (bf16_t)a0.w;
  a[4]  = (bf16_t)a1.x; a[5]  = (bf16_t)a1.y; a[6]  = (bf16_t)a1.z; a[7]  = (bf16_t)a1.w;
  a[8]  = (bf16_t)a2.x; a[9]  = (bf16_t)a2.y; a[10] = (bf16_t)a2.z; a[11] = (bf16_t)a2.w;
  a[12] = (bf16_t)a3.x; a[13] = (bf16_t)a3.y; a[14] = (bf16_t)a3.z; a[15] = (bf16_t)a3.w;
  return a;
}

// Shared inner body: acc += A[16, k_begin:k_begin+krange] * W[..., col-slice]
__device__ __forceinline__ v8f gemm_slice(const float* __restrict__ arow,
                                          const float* __restrict__ W,
                                          int k_begin, int krange,
                                          int col0, int kg, int col) {
  v8f acc = {};
  for (int k0 = k_begin; k0 < k_begin + krange; k0 += 32) {
    v16bf a = make_a_frag(arow, k0, kg);
    // B-fragment (32x16): lane holds column `col`, K = kg*16 + i  (contiguous order, same as the A-fragment)
    const float* __restrict__ wcol =
        W + (size_t)(k0 + kg * 16) * DIM + col0 + col;
    if (k0 + 32 < k_begin + krange)
      __builtin_prefetch(wcol + (size_t)32 * DIM, 0, 1);  // global_prefetch_b8
    v16bf bfrag;
#pragma unroll
    for (int i = 0; i < 16; ++i) bfrag[i] = (bf16_t)wcol[(size_t)i * DIM];
    acc = __builtin_amdgcn_wmma_f32_16x16x32_bf16(
        false, a, false, bfrag, (short)0, acc, false, false);
  }
  return acc;
}

// ---------------------------------------------------------------------------
// QKV projection: part[mat][ks][16][DIM] = X[16,Krange] @ W[Krange, DIM-slice]
// grid = (DIM/128, KSPLIT, 3), block = 256 (8 waves; each wave owns 16 cols)
// ---------------------------------------------------------------------------
__global__ void __launch_bounds__(256)
qkv_gemm(const float* __restrict__ x,
         const float* __restrict__ wq,
         const float* __restrict__ wk,
         const float* __restrict__ wv,
         float* __restrict__ part) {
  const int wave = threadIdx.x >> 5;
  const int lane = threadIdx.x & 31;
  const int col0 = blockIdx.x * 128 + wave * 16;
  const int ks   = blockIdx.y;
  const int mat  = blockIdx.z;
  const float* __restrict__ W = (mat == 0) ? wq : (mat == 1) ? wk : wv;

  const int row = lane & 15;   // A: M index
  const int kg  = lane >> 4;   // A/B K-group select
  const int col = lane & 15;   // B/C: N index

  v8f acc = gemm_slice(x + (size_t)row * DIM, W, ks * KRANGE, KRANGE,
                       col0, kg, col);

  // C/D layout: VGPR v -> M = (lane>>4)*8 + v, N = lane&15
  float* __restrict__ outp = part + (((size_t)mat * KSPLIT + ks) * BSZ) * DIM;
  const int mbase = (lane >> 4) * 8;
#pragma unroll
  for (int v = 0; v < 8; ++v)
    outp[(size_t)(mbase + v) * DIM + col0 + col] = acc[v];
}

// ---------------------------------------------------------------------------
// Reduce K-split partials + rotary (q,k) ; v passthrough.
// One thread per (batch, even/odd pair). grid = 16*2048/256 blocks.
// ---------------------------------------------------------------------------
__global__ void __launch_bounds__(256)
reduce_rotary(const float* __restrict__ part,
              const float* __restrict__ fcos,
              const float* __restrict__ fsin,
              float* __restrict__ xq,
              float* __restrict__ xk,
              float* __restrict__ xv) {
  int idx = blockIdx.x * blockDim.x + threadIdx.x;
  if (idx >= BSZ * (DIM / 2)) return;
  int b  = idx / (DIM / 2);
  int p  = idx % (DIM / 2);
  int d0 = p * 2;
  int j  = (d0 & (HD - 1)) >> 1;           // rotary pair index within head
  size_t base = (size_t)b * DIM + d0;

  float qe = 0.f, qo = 0.f, ke = 0.f, ko = 0.f, ve = 0.f, vo = 0.f;
#pragma unroll
  for (int s = 0; s < KSPLIT; ++s) {
    const float* pq = part + (((size_t)0 * KSPLIT + s) * BSZ) * DIM;
    const float* pk = part + (((size_t)1 * KSPLIT + s) * BSZ) * DIM;
    const float* pv = part + (((size_t)2 * KSPLIT + s) * BSZ) * DIM;
    qe += pq[base]; qo += pq[base + 1];
    ke += pk[base]; ko += pk[base + 1];
    ve += pv[base]; vo += pv[base + 1];
  }
  float c = fcos[j], sn = fsin[j];
  xq[base]     = qe * c - qo * sn;
  xq[base + 1] = qe * sn + qo * c;
  xk[base]     = ke * c - ko * sn;
  xk[base + 1] = ke * sn + ko * c;
  xv[base]     = ve;
  xv[base + 1] = vo;
}

// ---------------------------------------------------------------------------
// Flash-decode partial: grid = (16*32, SPLITS), block = 256 (8 waves).
// Each wave streams whole 512B K/V rows (float4 per lane) -> fully coalesced.
// New key/value at s == NEWPOS is taken from workspace (inputs not mutated).
// ---------------------------------------------------------------------------
__global__ void __launch_bounds__(256)
attn_partial(const float* __restrict__ cache_k,
             const float* __restrict__ cache_v,
             const float* __restrict__ xq,
             const float* __restrict__ xk,
             const float* __restrict__ xv,
             float* __restrict__ parts) {
  __shared__ float sProb[CHUNK];
  __shared__ float sRed[16];
  __shared__ float sAcc[8][HD];

  const int bh    = blockIdx.x;
  const int b     = bh >> 5;
  const int h     = bh & 31;
  const int split = blockIdx.y;
  const int wave  = threadIdx.x >> 5;
  const int lane  = threadIdx.x & 31;
  const int s_begin = split * CHUNK;

  const float scale = 0.08838834764831845f;  // 1/sqrt(128)
  const size_t qoff = (size_t)b * DIM + (size_t)h * HD + lane * 4;
  float4 q = *reinterpret_cast<const float4*>(xq + qoff);
  q.x *= scale; q.y *= scale; q.z *= scale; q.w *= scale;

  // -- scores: one K row per wave per iteration
  for (int s = s_begin + wave; s < s_begin + CHUNK; s += 8) {
    const float* krow = (s == NEWPOS)
        ? (xk + (size_t)b * DIM + (size_t)h * HD)
        : (cache_k + (((size_t)b * MAXSEQ + s) * NH + h) * HD);
    float4 kv = *reinterpret_cast<const float4*>(krow + lane * 4);
    float d = q.x * kv.x + q.y * kv.y + q.z * kv.z + q.w * kv.w;
#pragma unroll
    for (int off = 16; off; off >>= 1) d += __shfl_xor(d, off, 32);
    if (lane == 0) sProb[s - s_begin] = d;
  }
  __syncthreads();

  // -- block max
  float m = -INFINITY;
  for (int i = threadIdx.x; i < CHUNK; i += 256) m = fmaxf(m, sProb[i]);
#pragma unroll
  for (int off = 16; off; off >>= 1) m = fmaxf(m, __shfl_xor(m, off, 32));
  if (lane == 0) sRed[wave] = m;
  __syncthreads();
  m = sRed[0];
#pragma unroll
  for (int w = 1; w < 8; ++w) m = fmaxf(m, sRed[w]);

  // -- exp + local sum (same thread reads/writes each sProb slot)
  float lsum = 0.f;
  for (int i = threadIdx.x; i < CHUNK; i += 256) {
    float pexp = __expf(sProb[i] - m);
    sProb[i] = pexp;
    lsum += pexp;
  }
#pragma unroll
  for (int off = 16; off; off >>= 1) lsum += __shfl_xor(lsum, off, 32);
  if (lane == 0) sRed[8 + wave] = lsum;
  __syncthreads();
  float L = 0.f;
#pragma unroll
  for (int w = 0; w < 8; ++w) L += sRed[8 + w];

  // -- P @ V for this chunk (unnormalized)
  float4 acc = {0.f, 0.f, 0.f, 0.f};
  for (int s = s_begin + wave; s < s_begin + CHUNK; s += 8) {
    const float* vrow = (s == NEWPOS)
        ? (xv + (size_t)b * DIM + (size_t)h * HD)
        : (cache_v + (((size_t)b * MAXSEQ + s) * NH + h) * HD);
    float4 vv = *reinterpret_cast<const float4*>(vrow + lane * 4);
    float pw = sProb[s - s_begin];
    acc.x += pw * vv.x; acc.y += pw * vv.y;
    acc.z += pw * vv.z; acc.w += pw * vv.w;
  }
  *reinterpret_cast<float4*>(&sAcc[wave][lane * 4]) = acc;
  __syncthreads();

  if (wave == 0) {
    float4 t = {0.f, 0.f, 0.f, 0.f};
#pragma unroll
    for (int w = 0; w < 8; ++w) {
      float4 u = *reinterpret_cast<float4*>(&sAcc[w][lane * 4]);
      t.x += u.x; t.y += u.y; t.z += u.z; t.w += u.w;
    }
    float* op = parts + ((size_t)bh * SPLITS + split) * PSTRIDE;
    *reinterpret_cast<float4*>(op + 4 + lane * 4) = t;
    if (lane == 0) { op[0] = m; op[1] = L; }
  }
}

// ---------------------------------------------------------------------------
// Merge SPLITS partials. grid = 512, block = 128 (one thread per d).
// ---------------------------------------------------------------------------
__global__ void __launch_bounds__(128)
attn_combine(const float* __restrict__ parts,
             float* __restrict__ attn_out) {
  const int bh = blockIdx.x;
  const int d  = threadIdx.x;
  const float* p = parts + (size_t)bh * SPLITS * PSTRIDE;
  float M = -INFINITY;
#pragma unroll
  for (int s = 0; s < SPLITS; ++s) M = fmaxf(M, p[(size_t)s * PSTRIDE]);
  float L = 0.f, acc = 0.f;
#pragma unroll
  for (int s = 0; s < SPLITS; ++s) {
    float w = __expf(p[(size_t)s * PSTRIDE] - M);
    L   += p[(size_t)s * PSTRIDE + 1] * w;
    acc += p[(size_t)s * PSTRIDE + 4 + d] * w;
  }
  attn_out[(size_t)bh * HD + d] = acc / L;  // [b][h][d] == reshape(h*d)
}

// ---------------------------------------------------------------------------
// Output projection, split-K: opart[ks][16][DIM] = attn_out @ wo[slice]
// grid = (DIM/128, OSPLIT), block = 256.
// ---------------------------------------------------------------------------
__global__ void __launch_bounds__(256)
out_gemm(const float* __restrict__ a_in,
         const float* __restrict__ wo,
         float* __restrict__ opart) {
  const int wave = threadIdx.x >> 5;
  const int lane = threadIdx.x & 31;
  const int col0 = blockIdx.x * 128 + wave * 16;
  const int ks   = blockIdx.y;
  const int row  = lane & 15;
  const int kg   = lane >> 4;
  const int col  = lane & 15;

  v8f acc = gemm_slice(a_in + (size_t)row * DIM, wo, ks * ORANGE, ORANGE,
                       col0, kg, col);

  float* __restrict__ outp = opart + ((size_t)ks * BSZ) * DIM;
  const int mbase = (lane >> 4) * 8;
#pragma unroll
  for (int v = 0; v < 8; ++v)
    outp[(size_t)(mbase + v) * DIM + col0 + col] = acc[v];
}

// ---------------------------------------------------------------------------
// Sum OSPLIT partials into d_out. grid = 16*4096/256.
// ---------------------------------------------------------------------------
__global__ void __launch_bounds__(256)
out_reduce(const float* __restrict__ opart, float* __restrict__ out) {
  int idx = blockIdx.x * blockDim.x + threadIdx.x;
  if (idx >= BSZ * DIM) return;
  float s = 0.f;
#pragma unroll
  for (int k = 0; k < OSPLIT; ++k)
    s += opart[(size_t)k * BSZ * DIM + idx];
  out[idx] = s;
}

// ---------------------------------------------------------------------------
extern "C" void kernel_launch(void* const* d_in, const int* in_sizes, int n_in,
                              void* d_out, int out_size, void* d_ws, size_t ws_size,
                              hipStream_t stream) {
  const float* x        = (const float*)d_in[0];
  const float* wq       = (const float*)d_in[1];
  const float* wk       = (const float*)d_in[2];
  const float* wv       = (const float*)d_in[3];
  const float* wo       = (const float*)d_in[4];
  const float* cache_k  = (const float*)d_in[5];
  const float* cache_v  = (const float*)d_in[6];
  const float* fcos     = (const float*)d_in[7];
  const float* fsin     = (const float*)d_in[8];
  float* out = (float*)d_out;

  // Workspace carve-out (floats)
  float* ws = (float*)d_ws;
  float* part_qkv   = ws;                                   // 3*KSPLIT*16*DIM
  float* xq         = part_qkv + (size_t)3 * KSPLIT * BSZ * DIM;
  float* xk         = xq + (size_t)BSZ * DIM;
  float* xv         = xk + (size_t)BSZ * DIM;
  float* attn_parts = xv + (size_t)BSZ * DIM;               // 512*SPLITS*PSTRIDE
  float* attn_out   = attn_parts + (size_t)BSZ * NH * SPLITS * PSTRIDE;
  float* opart      = attn_out + (size_t)BSZ * DIM;         // OSPLIT*16*DIM

  // 1) Q/K/V projections (WMMA bf16, K split 8 ways -> 768 WGs)
  qkv_gemm<<<dim3(DIM / 128, KSPLIT, 3), 256, 0, stream>>>(x, wq, wk, wv, part_qkv);

  // 2) K-split reduce + rotary
  {
    int n = BSZ * (DIM / 2);
    reduce_rotary<<<(n + 255) / 256, 256, 0, stream>>>(part_qkv, fcos, fsin, xq, xk, xv);
  }

  // 3) Flash-decode partials over the KV cache (bandwidth-dominant pass)
  attn_partial<<<dim3(BSZ * NH, SPLITS), 256, 0, stream>>>(
      cache_k, cache_v, xq, xk, xv, attn_parts);

  // 4) Merge splits
  attn_combine<<<BSZ * NH, HD, 0, stream>>>(attn_parts, attn_out);

  // 5) Output projection (WMMA bf16, K split 8 ways -> 256 WGs)
  out_gemm<<<dim3(DIM / 128, OSPLIT), 256, 0, stream>>>(attn_out, wo, opart);

  // 6) Sum output partials into d_out
  {
    int n = BSZ * DIM;
    out_reduce<<<(n + 255) / 256, 256, 0, stream>>>(opart, out);
  }
}
